// CrossAttention_46050639347816
// MI455X (gfx1250) — matrix operand order repaired and checked by reading, not compiled, not run on hardware
//
#include <hip/hip_runtime.h>
#include <hip/hip_bf16.h>

// Problem constants (match reference)
#define B_   4
#define N_   1024
#define C_   1024
#define H_   16
#define D_   64
#define BH_  (B_ * H_)          // 64
#define M_   (B_ * N_)          // 4096
#define K_RANK 104858           // numel - max(1,int(0.9*numel)) for N*N=1048576
#define QK_SCALE 0.125f         // D^-0.5
#define SLICES 16               // scan parallelism per (b,h) in radix select

typedef __bf16 bf16;
typedef __attribute__((ext_vector_type(16))) __bf16 v16bf;
typedef __attribute__((ext_vector_type(8)))  __bf16 v8bf;
typedef __attribute__((ext_vector_type(8)))  float  v8f;

// ---------------------------------------------------------------------------
// WMMA fragment helpers (CDNA5 wave32 layouts, 16x16x32 bf16)
// ---------------------------------------------------------------------------

// A-matrix 16x32 (MxK), row-major source with leading dim `ld`.
// lanes 0-15 : row = lane,    K = 0..7 and 16..23
// lanes 16-31: row = lane-16, K = 8..15 and 24..31
__device__ __forceinline__ v16bf load_a_frag(const bf16* __restrict__ base, int ld) {
    int lane  = threadIdx.x & 31;
    int row   = lane & 15;
    int khalf = lane >> 4;
    const bf16* p = base + (size_t)row * ld + khalf * 8;
    v8bf lo = *(const v8bf*)(p);        // K = khalf*8 .. +7
    v8bf hi = *(const v8bf*)(p + 16);   // K = khalf*8+16 .. +23
    v16bf r;
#pragma unroll
    for (int i = 0; i < 8; ++i) { r[i] = lo[i]; r[8 + i] = hi[i]; }
    return r;
}

// B-matrix 32x16 (KxN). Source `w` stores B^T row-major: w[col*ld + k] = B[k][col]
// (weight rows / K rows / V^T rows are B columns -> fully contiguous loads).
__device__ __forceinline__ v16bf load_b_frag(const bf16* __restrict__ w, int ld,
                                             int n0, int k0) {
    int lane = threadIdx.x & 31;
    int col  = n0 + (lane & 15);
    int ks   = k0 + (lane >> 4) * 8;
    const bf16* p = w + (size_t)col * ld + ks;
    v8bf a = *(const v8bf*)(p);
    v8bf b = *(const v8bf*)(p + 16);
    v16bf r;
#pragma unroll
    for (int i = 0; i < 8; ++i) { r[i] = a[i]; r[8 + i] = b[i]; }
    return r;
}

__device__ __forceinline__ v8f wmma_bf16(v16bf a, v16bf b, v8f c) {
    return __builtin_amdgcn_wmma_f32_16x16x32_bf16(false, a, false, b,
                                                   (short)0, c, false, false);
}

// ---------------------------------------------------------------------------
// Block reduction helpers
// ---------------------------------------------------------------------------
__device__ __forceinline__ float block_sum(float v, float* sm) {
    sm[threadIdx.x] = v; __syncthreads();
    for (int s = blockDim.x >> 1; s > 0; s >>= 1) {
        if (threadIdx.x < s) sm[threadIdx.x] += sm[threadIdx.x + s];
        __syncthreads();
    }
    float r = sm[0]; __syncthreads();
    return r;
}
__device__ __forceinline__ float block_max(float v, float* sm) {
    sm[threadIdx.x] = v; __syncthreads();
    for (int s = blockDim.x >> 1; s > 0; s >>= 1) {
        if (threadIdx.x < s) sm[threadIdx.x] = fmaxf(sm[threadIdx.x], sm[threadIdx.x + s]);
        __syncthreads();
    }
    float r = sm[0]; __syncthreads();
    return r;
}

// ---------------------------------------------------------------------------
// 1) LayerNorm -> bf16    (one block = one row of C)
// ---------------------------------------------------------------------------
__global__ void ln_bf16_kernel(const float* __restrict__ x, const float* __restrict__ g,
                               const float* __restrict__ be, bf16* __restrict__ out) {
    __shared__ float sm[256];
    int row = blockIdx.x;
    const float* xr = x + (size_t)row * C_;
    float v[4], lsum = 0.f, lsq = 0.f;
#pragma unroll
    for (int j = 0; j < 4; ++j) {
        v[j] = xr[threadIdx.x + j * 256];
        lsum += v[j]; lsq += v[j] * v[j];
    }
    float mu  = block_sum(lsum, sm) * (1.0f / C_);
    float var = block_sum(lsq, sm) * (1.0f / C_) - mu * mu;
    float inv = rsqrtf(var + 1e-5f);
#pragma unroll
    for (int j = 0; j < 4; ++j) {
        int c = threadIdx.x + j * 256;
        out[(size_t)row * C_ + c] = (bf16)((v[j] - mu) * inv * g[c] + be[c]);
    }
}

// ---------------------------------------------------------------------------
// 2) f32 -> bf16 conversion (weights)
// ---------------------------------------------------------------------------
__global__ void cvt_bf16_kernel(const float* __restrict__ src, bf16* __restrict__ dst, int n) {
    for (int i = blockIdx.x * blockDim.x + threadIdx.x; i < n; i += gridDim.x * blockDim.x)
        dst[i] = (bf16)src[i];
}

// ---------------------------------------------------------------------------
// 3) Projection GEMM: out = A[M x K] @ W^T + bias, bf16 WMMA, f32 accum.
//    mode 0: bf16 out in head layout   [B,H,N,D]
//    mode 2: bf16 out transposed heads [B,H,D,N]   (for V)
//    mode 1: f32 out row-major [M x Nout]          (final projection)
// ---------------------------------------------------------------------------
__device__ __forceinline__ void store_tile(v8f acc, int m0, int c0,
                                           const float* __restrict__ bias,
                                           bf16* __restrict__ out_bf,
                                           float* __restrict__ out_f,
                                           int mode, int Nout) {
    int lane = threadIdx.x & 31;
    int hi   = lane >> 4;
    int col  = c0 + (lane & 15);
    float bv = bias[col];
#pragma unroll
    for (int j = 0; j < 8; ++j) {
        int gm   = m0 + j + hi * 8;
        float v  = acc[j] + bv;
        if (mode == 1) {
            out_f[(size_t)gm * Nout + col] = v;
        } else {
            int b = gm >> 10, n = gm & (N_ - 1);
            int h = col >> 6, d = col & (D_ - 1);
            size_t idx = (mode == 0)
                ? ((((size_t)(b * H_ + h)) * N_ + n) * D_ + d)
                : ((((size_t)(b * H_ + h)) * D_ + d) * N_ + n);
            out_bf[idx] = (bf16)v;
        }
    }
}

__global__ void proj_gemm_kernel(const bf16* __restrict__ A, const bf16* __restrict__ W,
                                 const float* __restrict__ bias,
                                 bf16* __restrict__ out_bf, float* __restrict__ out_f,
                                 int mode) {
    const int K = C_, Nout = C_;
    int wave = threadIdx.x >> 5;
    int m0   = blockIdx.x * 16;
    int c0   = blockIdx.y * 256 + wave * 32;
    v8f acc0 = {}; v8f acc1 = {};
    for (int k0 = 0; k0 < K; k0 += 32) {
        __builtin_prefetch(A + (size_t)m0 * K + k0 + 64, 0, 1);
        v16bf a  = load_a_frag(A + (size_t)m0 * K + k0, K);
        v16bf b0 = load_b_frag(W, K, c0,      k0);
        v16bf b1 = load_b_frag(W, K, c0 + 16, k0);
        acc0 = wmma_bf16(a, b0, acc0);
        acc1 = wmma_bf16(a, b1, acc1);
    }
    store_tile(acc0, m0, c0,      bias, out_bf, out_f, mode, Nout);
    store_tile(acc1, m0, c0 + 16, bias, out_bf, out_f, mode, Nout);
}

// ---------------------------------------------------------------------------
// 4) Scores: S[bh][n][m] = scale * Q[bh][n][:] . K[bh][m][:]
// ---------------------------------------------------------------------------
__global__ void qk_scores_kernel(const bf16* __restrict__ qh, const bf16* __restrict__ kh,
                                 float* __restrict__ attn) {
    int bh   = blockIdx.y;
    int wave = threadIdx.x >> 5;
    int m0   = blockIdx.x * 16;
    const bf16* qb = qh + ((size_t)bh * N_ + m0) * D_;
    const bf16* kb = kh + (size_t)bh * N_ * D_;
    v16bf a0 = load_a_frag(qb,      D_);     // K-dim 0..31
    v16bf a1 = load_a_frag(qb + 32, D_);     // K-dim 32..63
    float* srow = attn + ((size_t)bh * N_ + m0) * N_;
    int lane = threadIdx.x & 31, hi = lane >> 4;
#pragma unroll
    for (int t = 0; t < 8; ++t) {
        int c0 = (wave * 8 + t) * 16;
        v16bf b0 = load_b_frag(kb, D_, c0, 0);
        v16bf b1 = load_b_frag(kb, D_, c0, 32);
        v8f acc = {};
        acc = wmma_bf16(a0, b0, acc);
        acc = wmma_bf16(a1, b1, acc);
        int col = c0 + (lane & 15);
#pragma unroll
        for (int j = 0; j < 8; ++j)
            srow[(size_t)(j + hi * 8) * N_ + col] = acc[j] * QK_SCALE;
    }
}

// ---------------------------------------------------------------------------
// 5) Exact kthvalue per (b,h): 4-pass radix select, multi-block scans.
//    Scan: grid (SLICES, BH); shared 256-bin partial histogram -> global.
//    Select: grid BH; picks bin, updates prefix/k, zeros hist for next pass.
// ---------------------------------------------------------------------------
__global__ void radix_init_kernel(unsigned* __restrict__ prefix, int* __restrict__ kk,
                                  unsigned* __restrict__ hist) {
    int i = blockIdx.x * blockDim.x + threadIdx.x;
    if (i < BH_) { prefix[i] = 0u; kk[i] = K_RANK; }
    for (int j = i; j < BH_ * 256; j += gridDim.x * blockDim.x) hist[j] = 0u;
}

__global__ void radix_scan_kernel(const float* __restrict__ attn,
                                  const unsigned* __restrict__ prefix,
                                  unsigned* __restrict__ hist, int pass) {
    __shared__ unsigned lh[256];
    int bh = blockIdx.y;
    lh[threadIdx.x] = 0u;
    __syncthreads();
    int shift = 24 - 8 * pass;
    unsigned pmask = (pass == 0) ? 0u : (0xFFFFFFFFu << (shift + 8));
    unsigned pfx = prefix[bh];
    const float* s = attn + (size_t)bh * N_ * N_;
    const int per = (N_ * N_) / SLICES;               // 65536
    int start = blockIdx.x * per;
    for (int i = start + threadIdx.x; i < start + per; i += 256) {
        unsigned bits = __float_as_uint(s[i]);
        unsigned key  = bits ^ ((bits & 0x80000000u) ? 0xFFFFFFFFu : 0x80000000u);
        if (pass == 0 || ((key & pmask) == pfx))
            atomicAdd(&lh[(key >> shift) & 255u], 1u);
    }
    __syncthreads();
    unsigned c = lh[threadIdx.x];
    if (c) atomicAdd(&hist[bh * 256 + threadIdx.x], c);
}

__global__ void radix_select_step(unsigned* __restrict__ hist, unsigned* __restrict__ prefix,
                                  int* __restrict__ kk, float* __restrict__ thr, int pass) {
    int bh = blockIdx.x;
    unsigned* h = hist + bh * 256;
    if (threadIdx.x == 0) {
        int k = kk[bh];
        unsigned cum = 0; int bin = 255;
        for (int b2 = 0; b2 < 256; ++b2) {
            if (cum + h[b2] >= (unsigned)k) { bin = b2; break; }
            cum += h[b2];
        }
        int shift = 24 - 8 * pass;
        unsigned pfx = prefix[bh] | ((unsigned)bin << shift);
        prefix[bh] = pfx;
        kk[bh] = k - (int)cum;
        if (pass == 3) {
            unsigned bits = (pfx & 0x80000000u) ? (pfx ^ 0x80000000u) : ~pfx;
            thr[bh] = __uint_as_float(bits);
        }
    }
    __syncthreads();
    for (int i = threadIdx.x; i < 256; i += blockDim.x) h[i] = 0u;  // ready for next pass
}

// ---------------------------------------------------------------------------
// 6) Mask + clip + softmax (in-place on attn) + bf16 copy of probs.
// ---------------------------------------------------------------------------
__global__ void mask_softmax_kernel(float* __restrict__ attn, const float* __restrict__ thr,
                                    bf16* __restrict__ probs) {
    __shared__ float sm[256];
    int n = blockIdx.x, bh = blockIdx.y;
    size_t base = ((size_t)bh * N_ + n) * N_;
    float t = thr[bh];
    float v[4], mx = -3.0e38f;
#pragma unroll
    for (int j = 0; j < 4; ++j) {
        float x = attn[base + threadIdx.x + j * 256];
        x = (x < t) ? -10000.0f : x;
        x = fminf(fmaxf(x, -10000.0f), 10000.0f);
        v[j] = x;
        mx = fmaxf(mx, x);
    }
    float rmax = block_max(mx, sm);
    float lsum = 0.f;
#pragma unroll
    for (int j = 0; j < 4; ++j) { v[j] = __expf(v[j] - rmax); lsum += v[j]; }
    float inv = 1.0f / block_sum(lsum, sm);
#pragma unroll
    for (int j = 0; j < 4; ++j) {
        size_t i = base + threadIdx.x + j * 256;
        float p = v[j] * inv;
        attn[i]  = p;
        probs[i] = (bf16)p;
    }
}

// ---------------------------------------------------------------------------
// 7) P.V: O[bh][n][d] = sum_m P[bh][n][m] * V[bh][m][d], V stored transposed.
// ---------------------------------------------------------------------------
__global__ void pv_gemm_kernel(const bf16* __restrict__ probs, const bf16* __restrict__ vT,
                               bf16* __restrict__ Obf) {
    int bh   = blockIdx.y;
    int wave = threadIdx.x >> 5;
    int m0   = (blockIdx.x * 8 + wave) * 16;
    const bf16* pb = probs + (size_t)bh * N_ * N_;
    const bf16* vb = vT    + (size_t)bh * D_ * N_;
    v8f acc[4] = {{}, {}, {}, {}};
    for (int k0 = 0; k0 < N_; k0 += 32) {
        __builtin_prefetch(pb + (size_t)m0 * N_ + k0 + 64, 0, 1);
        v16bf a = load_a_frag(pb + (size_t)m0 * N_ + k0, N_);
#pragma unroll
        for (int t = 0; t < 4; ++t) {
            v16bf b = load_b_frag(vb, N_, t * 16, k0);
            acc[t] = wmma_bf16(a, b, acc[t]);
        }
    }
    int b = bh >> 4, h = bh & 15;
    int lane = threadIdx.x & 31, hi = lane >> 4;
#pragma unroll
    for (int t = 0; t < 4; ++t) {
        int col = h * D_ + t * 16 + (lane & 15);
#pragma unroll
        for (int j = 0; j < 8; ++j) {
            int n = m0 + j + hi * 8;
            Obf[((size_t)(b * N_ + n)) * C_ + col] = (bf16)acc[t][j];
        }
    }
}

// ---------------------------------------------------------------------------
// 8) Residual + final LayerNorm -> d_out x region (f32)
// ---------------------------------------------------------------------------
__global__ void residual_ln_kernel(const float* __restrict__ x1, const float* __restrict__ ident,
                                   const float* __restrict__ g, const float* __restrict__ be,
                                   float* __restrict__ out) {
    __shared__ float sm[256];
    int row = blockIdx.x;
    size_t rb = (size_t)row * C_;
    float v[4], lsum = 0.f, lsq = 0.f;
#pragma unroll
    for (int j = 0; j < 4; ++j) {
        int c = threadIdx.x + j * 256;
        v[j] = x1[rb + c] + ident[rb + c];
        lsum += v[j]; lsq += v[j] * v[j];
    }
    float mu  = block_sum(lsum, sm) * (1.0f / C_);
    float var = block_sum(lsq, sm) * (1.0f / C_) - mu * mu;
    float inv = rsqrtf(var + 1e-5f);
#pragma unroll
    for (int j = 0; j < 4; ++j) {
        int c = threadIdx.x + j * 256;
        out[rb + c] = (v[j] - mu) * inv * g[c] + be[c];
    }
}

// ---------------------------------------------------------------------------
// Host launcher
// ---------------------------------------------------------------------------
extern "C" void kernel_launch(void* const* d_in, const int* in_sizes, int n_in,
                              void* d_out, int out_size, void* d_ws, size_t ws_size,
                              hipStream_t stream) {
    const float* q_in   = (const float*)d_in[0];
    const float* kv_in  = (const float*)d_in[1];
    const float* g_q    = (const float*)d_in[2];
    const float* b_q    = (const float*)d_in[3];
    const float* g_k    = (const float*)d_in[4];
    const float* b_k    = (const float*)d_in[5];
    const float* g_v    = (const float*)d_in[6];
    const float* b_v    = (const float*)d_in[7];
    const float* Wq     = (const float*)d_in[8];
    const float* bq     = (const float*)d_in[9];
    const float* Wk     = (const float*)d_in[10];
    const float* bk     = (const float*)d_in[11];
    const float* Wv     = (const float*)d_in[12];
    const float* bv     = (const float*)d_in[13];
    const float* Wo     = (const float*)d_in[14];
    const float* bo     = (const float*)d_in[15];
    const float* g_out  = (const float*)d_in[16];
    const float* b_out  = (const float*)d_in[17];

    float* out_x    = (float*)d_out;                       // [B,N,C]
    float* out_attn = out_x + (size_t)B_ * N_ * C_;        // [B,H,N,N]

    // Workspace layout
    char* ws = (char*)d_ws;
    const size_t SZ_ACT   = (size_t)M_ * C_ * sizeof(bf16);       // 8 MB
    const size_t SZ_W     = (size_t)C_ * C_ * sizeof(bf16);       // 2 MB
    const size_t SZ_PROBS = (size_t)BH_ * N_ * N_ * sizeof(bf16); // 128 MB
    const size_t SZ_X1    = (size_t)M_ * C_ * sizeof(float);      // 16 MB
    size_t off = 0;
    bf16* qn   = (bf16*)(ws + off); off += SZ_ACT;
    bf16* kn   = (bf16*)(ws + off); off += SZ_ACT;
    bf16* vn   = (bf16*)(ws + off); off += SZ_ACT;
    bf16* Wqb  = (bf16*)(ws + off); off += SZ_W;
    bf16* Wkb  = (bf16*)(ws + off); off += SZ_W;
    bf16* Wvb  = (bf16*)(ws + off); off += SZ_W;
    bf16* Wob  = (bf16*)(ws + off); off += SZ_W;
    bf16* qh   = (bf16*)(ws + off); off += SZ_ACT;   // [B,H,N,D]
    bf16* kh   = (bf16*)(ws + off); off += SZ_ACT;   // [B,H,N,D]
    bf16* vT   = (bf16*)(ws + off); off += SZ_ACT;   // [B,H,D,N]
    bf16* prob = (bf16*)(ws + off); off += SZ_PROBS;
    bf16* Obf  = (bf16*)(ws + off); off += SZ_ACT;   // [B,N,C] bf16
    float* x1  = (float*)(ws + off); off += SZ_X1;
    float* thr = (float*)(ws + off); off += 256 * sizeof(float);
    unsigned* r_prefix = (unsigned*)(ws + off); off += 256 * sizeof(unsigned);
    int*      r_k      = (int*)(ws + off);      off += 256 * sizeof(int);
    unsigned* r_hist   = (unsigned*)(ws + off); off += (size_t)BH_ * 256 * sizeof(unsigned);

    // 1) LayerNorms -> bf16
    ln_bf16_kernel<<<M_, 256, 0, stream>>>(q_in,  g_q, b_q, qn);
    ln_bf16_kernel<<<M_, 256, 0, stream>>>(kv_in, g_k, b_k, kn);
    ln_bf16_kernel<<<M_, 256, 0, stream>>>(kv_in, g_v, b_v, vn);

    // 2) Weights -> bf16
    cvt_bf16_kernel<<<512, 256, 0, stream>>>(Wq, Wqb, C_ * C_);
    cvt_bf16_kernel<<<512, 256, 0, stream>>>(Wk, Wkb, C_ * C_);
    cvt_bf16_kernel<<<512, 256, 0, stream>>>(Wv, Wvb, C_ * C_);
    cvt_bf16_kernel<<<512, 256, 0, stream>>>(Wo, Wob, C_ * C_);

    // 3) Q/K/V projections (WMMA)
    dim3 pgrid(M_ / 16, C_ / 256);
    proj_gemm_kernel<<<pgrid, 256, 0, stream>>>(qn, Wqb, bq, qh, nullptr, 0);
    proj_gemm_kernel<<<pgrid, 256, 0, stream>>>(kn, Wkb, bk, kh, nullptr, 0);
    proj_gemm_kernel<<<pgrid, 256, 0, stream>>>(vn, Wvb, bv, vT, nullptr, 2);

    // 4) Scaled scores into the attn output region (WMMA)
    qk_scores_kernel<<<dim3(N_ / 16, BH_), 256, 0, stream>>>(qh, kh, out_attn);

    // 5) Exact per-(b,h) kthvalue threshold: parallel 4-pass radix select
    radix_init_kernel<<<64, 256, 0, stream>>>(r_prefix, r_k, r_hist);
    for (int pass = 0; pass < 4; ++pass) {
        radix_scan_kernel<<<dim3(SLICES, BH_), 256, 0, stream>>>(out_attn, r_prefix, r_hist, pass);
        radix_select_step<<<BH_, 256, 0, stream>>>(r_hist, r_prefix, r_k, thr, pass);
    }

    // 6) Mask + clip + softmax in place, bf16 probs for PV
    mask_softmax_kernel<<<dim3(N_, BH_), 256, 0, stream>>>(out_attn, thr, prob);

    // 7) P.V (WMMA) -> bf16 [B,N,C]
    pv_gemm_kernel<<<dim3(N_ / 128, BH_), 256, 0, stream>>>(prob, vT, Obf);

    // 8) Output projection (WMMA, f32 out)
    proj_gemm_kernel<<<pgrid, 256, 0, stream>>>(Obf, Wob, bo, nullptr, x1, 1);

    // 9) Residual + final LN -> x output
    residual_ln_kernel<<<M_, 256, 0, stream>>>(x1, q_in, g_out, b_out, out_x);

    (void)in_sizes; (void)n_in; (void)out_size; (void)ws_size;
}
